// SFTL_10660108829253
// MI455X (gfx1250) — matrix operand order repaired and checked by reading, not compiled, not run on hardware
//
#include <hip/hip_runtime.h>
#include <hip/hip_bf16.h>
#include <math.h>

#define DEVI __device__ __forceinline__

typedef __attribute__((ext_vector_type(16))) __bf16 v16bf;
typedef __attribute__((ext_vector_type(8)))  float  v8f;

constexpr int  Bc = 2, DIMc = 64, Hc = 192, Wc = 192;
constexpr int  HWc = Hc * Wc;                      // 36864
constexpr int  HALFc = 32, HEADSc = 4, HDc = 8, KSc = 7;
constexpr long N64 = (long)Bc * DIMc * HWc;        // 4718592 floats
constexpr long N32 = (long)Bc * HALFc * HWc;       // 2359296 floats

// A-fragment K mapping for 16-bit WMMA (16x32 A): ISA 7.12.2
DEVI int kmapA(int e, int half) { return half * 8 + (e & 7) + ((e >> 3) << 4); }

DEVI v8f wmma_bf16(v16bf a, v16bf b, v8f c) {
  // v_wmma_f32_16x16x32_bf16  (neg_a, A, neg_b, B, c_mod, C, reuse_a, reuse_b)
  return __builtin_amdgcn_wmma_f32_16x16x32_bf16(false, a, false, b, (short)0, c, false, false);
}

// Pack 16 f32 (four float4 runs) into a bf16 fragment.
DEVI v16bf pack16(float4 a, float4 b, float4 c, float4 d) {
  v16bf r;
  r[0]=(__bf16)a.x; r[1]=(__bf16)a.y; r[2]=(__bf16)a.z; r[3]=(__bf16)a.w;
  r[4]=(__bf16)b.x; r[5]=(__bf16)b.y; r[6]=(__bf16)b.z; r[7]=(__bf16)b.w;
  r[8]=(__bf16)c.x; r[9]=(__bf16)c.y; r[10]=(__bf16)c.z; r[11]=(__bf16)c.w;
  r[12]=(__bf16)d.x; r[13]=(__bf16)d.y; r[14]=(__bf16)d.z; r[15]=(__bf16)d.w;
  return r;
}

// Load an A fragment from a 32-float row segment (rowPtr must be 16B aligned).
// Element e maps to k = kmapA(e, half): two contiguous 8-float runs.
DEVI v16bf load_a_row(const float* rowPtr, int half) {
  const float4* q = (const float4*)rowPtr;
  return pack16(q[half * 2], q[half * 2 + 1], q[half * 2 + 4], q[half * 2 + 5]);
}

// ---------------------------------------------------------------------------
// Channel LayerNorm over C channels in (B, C, H, W)-style layout.
// ---------------------------------------------------------------------------
__global__ void channel_ln_kernel(const float* __restrict__ src, float* __restrict__ dst,
                                  const float* __restrict__ g, const float* __restrict__ bv,
                                  int C, long srcBatch, int srcOff, long dstBatch, int tokens)
{
  int t = blockIdx.x * blockDim.x + threadIdx.x;
  if (t >= tokens) return;
  int b = t / HWc, hw = t % HWc;
  const float* sp = src + (long)b * srcBatch + (long)srcOff * HWc + hw;
  float mu = 0.f, ss = 0.f;
  for (int c = 0; c < C; ++c) { float v = sp[(long)c * HWc]; mu += v; ss += v * v; }
  mu /= (float)C;
  float var = ss / (float)C - mu * mu;
  float inv = rsqrtf(var + 1e-5f);
  float* dp = dst + (long)b * dstBatch + hw;
  for (int c = 0; c < C; ++c)
    dp[(long)c * HWc] = (sp[(long)c * HWc] - mu) * inv * g[c] + bv[c];
}

// ---------------------------------------------------------------------------
// Generic WMMA GEMM: C[b,m,n] = act( A[m,:]·B[b,:,n] + bias[m] ) + res
// A is (M,K) f32 row-major, B is (B, K, N) channel-major tokens, N = HW.
// A fragments (shared by all 8 waves of the block) staged once in LDS.
// ---------------------------------------------------------------------------
__global__ void gemm_wmma_kernel(const float* __restrict__ A, int lda,
                                 const float* __restrict__ Bm, long bBatch,
                                 const float* __restrict__ bias,
                                 const float* __restrict__ res, long resBatch, int resOff,
                                 float* __restrict__ C, long cBatch,
                                 int M, int N, int K, int act)
{
  __shared__ v16bf sA[4 * 32];                       // up to K=128 (4 chunks)
  const int lane = threadIdx.x & 31;
  const int wave = threadIdx.x >> 5;
  const int nw   = blockDim.x >> 5;
  const int idx = lane & 15, half = lane >> 4;
  int nb = (blockIdx.x * nw + wave) * 16;
  int mb = blockIdx.y * 16;
  int b  = blockIdx.z;
  const int KC = K >> 5;
  // ---- stage shared A fragments (identical for every wave in this block)
  for (int f = wave; f < KC; f += nw)
    sA[f * 32 + lane] = load_a_row(A + (long)(mb + idx) * lda + f * 32, half);
  __syncthreads();
  if (nb >= N) return;
  const float* Bp = Bm + (long)b * bBatch;
  v8f acc = {};
  for (int c = 0; c < KC; ++c) {
    int kb = c * 32;
    if (c + 1 < KC)            // prefetch next streamed B chunk into caches
      __builtin_prefetch(Bp + (long)(kb + 32 + half * 16) * N + nb + idx, 0, 1);
    v16bf af = sA[c * 32 + lane];
    v16bf bf;
#pragma unroll
    for (int e = 0; e < 16; ++e)
      bf[e] = (__bf16)Bp[(long)(kb + kmapA(e, half)) * N + nb + idx];
    acc = wmma_bf16(af, bf, acc);
  }
#pragma unroll
  for (int v = 0; v < 8; ++v) {
    int m = mb + half * 8 + v;
    int n = nb + idx;
    float val = acc[v] + bias[m];
    if (act == 1) val = 0.5f * val * (1.0f + erff(val * 0.70710678118654752f));
    if (res) val += res[(long)b * resBatch + (long)(resOff + m) * N + n];
    C[(long)b * cBatch + (long)m * N + n] = val;
  }
}

// ---------------------------------------------------------------------------
// 3x3 SAME conv as per-tap implicit GEMM via WMMA.
// X: (B, IC, H, W), W: (OC, IC, 3, 3), Y: (B, OC, H, W).
// Weight fragments (9 taps x IC/32 chunks, shared by the whole block) staged
// in LDS once; input patches streamed from global.
// ---------------------------------------------------------------------------
template<int IC>
__global__ void conv3x3_wmma_kernel(const float* __restrict__ X, long xBatch,
                                    const float* __restrict__ Wt,
                                    const float* __restrict__ bias,
                                    float* __restrict__ Y, long yBatch)
{
  constexpr int NCH = IC / 32;
  __shared__ v16bf sW[9 * NCH * 32];
  const int lane = threadIdx.x & 31, wave = threadIdx.x >> 5;
  const int nw = blockDim.x >> 5;
  const int idx = lane & 15, half = lane >> 4;
  int tile = blockIdx.x * nw + wave;
  int mb = blockIdx.y * 16;
  int b  = blockIdx.z;
  // ---- stage weight fragments (strided gather; done once per block)
  for (int f = wave; f < 9 * NCH; f += nw) {
    int tap = f / NCH, chunk = f % NCH;
    int kh = tap / 3, kw = tap % 3;
    v16bf a;
#pragma unroll
    for (int e = 0; e < 16; ++e) {
      int k = chunk * 32 + kmapA(e, half);
      a[e] = (__bf16)Wt[(((mb + idx) * IC + k) * 3 + kh) * 3 + kw];
    }
    sW[f * 32 + lane] = a;
  }
  __syncthreads();
  if (tile >= Hc * (Wc / 16)) return;
  int row = tile / (Wc / 16);
  int wst = (tile % (Wc / 16)) * 16;
  const float* Xb = X + (long)b * xBatch;
  v8f acc = {};
#pragma unroll
  for (int kh = 0; kh < 3; ++kh) {
    int hh = row + kh - 1;
    bool rok = (hh >= 0) && (hh < Hc);
#pragma unroll
    for (int kw = 0; kw < 3; ++kw) {
      int col = wst + idx + kw - 1;
      bool ok = rok && (col >= 0) && (col < Wc);
      long pix = (long)hh * Wc + col;
#pragma unroll
      for (int chunk = 0; chunk < NCH; ++chunk) {
        v16bf af = sW[((kh * 3 + kw) * NCH + chunk) * 32 + lane];
        v16bf bf;
#pragma unroll
        for (int e = 0; e < 16; ++e) {
          int k = chunk * 32 + kmapA(e, half);
          bf[e] = (__bf16)(ok ? Xb[(long)k * HWc + pix] : 0.0f);
        }
        acc = wmma_bf16(af, bf, acc);
      }
    }
  }
#pragma unroll
  for (int v = 0; v < 8; ++v) {
    int m = mb + half * 8 + v;
    Y[(long)b * yBatch + (long)m * HWc + (long)row * Wc + wst + idx] = acc[v] + bias[m];
  }
}

// ---------------------------------------------------------------------------
// 192-point DFT along rows (alongCols=0, stride-1 dim) or columns (alongCols=1)
// of 192x192 planes, as a 192x192 GEMM on WMMA. Twiddle fragments (identical
// for every wave in the block) are computed once and staged in LDS.
// Xi==nullptr => real input. Yi==nullptr => real-only output.
// ---------------------------------------------------------------------------
__global__ void dft192_wmma_kernel(const float* __restrict__ Xr, const float* __restrict__ Xi,
                                   float* __restrict__ Yr, float* __restrict__ Yi,
                                   int csel, long inBatch, float sign, float scale,
                                   int nPlanes, int alongCols)
{
  __shared__ v16bf sT[6 * 2 * 32];                  // [chunk][re/im][lane]
  const int lane = threadIdx.x & 31, wave = threadIdx.x >> 5;
  const int nw = blockDim.x >> 5;
  const int idx = lane & 15, half = lane >> 4;
  long tile = blockIdx.x * nw + wave;
  long Ntot = (long)nPlanes * 192;
  long nb = tile * 16;
  int mrow = blockIdx.y * 16 + idx;                 // A-row frequency (per-lane)
  float coef = sign * (6.283185307179586f / 192.0f) * (float)mrow;
  // ---- stage twiddle fragments once per block (8x less transcendental work)
  for (int c6 = wave; c6 < 6; c6 += nw) {
    int kb = c6 * 32;
    v16bf ar, ai;
#pragma unroll
    for (int e = 0; e < 16; ++e) {
      int j = kb + kmapA(e, half);
      float s, co;
      __sincosf(coef * (float)j, &s, &co);
      ar[e] = (__bf16)co; ai[e] = (__bf16)s;
    }
    sT[(c6 * 2 + 0) * 32 + lane] = ar;
    sT[(c6 * 2 + 1) * 32 + lane] = ai;
  }
  __syncthreads();
  if (nb >= Ntot) return;
  int p  = (int)(nb / 192);
  int nn = (int)(nb % 192) + idx;                   // non-transform index
  long inBase  = (long)(p / csel) * inBatch + (long)(p % csel) * HWc;
  long outBase = (long)p * HWc;
  bool cplx   = (Xi != nullptr);
  bool wantIm = (Yi != nullptr);
  v8f accR = {}, accI = {};
  for (int c6 = 0; c6 < 6; ++c6) {
    int kb = c6 * 32;
    v16bf ar = sT[(c6 * 2 + 0) * 32 + lane];
    v16bf ai = sT[(c6 * 2 + 1) * 32 + lane];
    v16bf br, bi, biN;
    if (!alongCols) {
      // B elements are two 8-float runs (k = kmapA): vectorized b128 loads
      const float* bp = Xr + inBase + (long)nn * 192 + kb;
      const float4* q = (const float4*)bp;
      br = pack16(q[half * 2], q[half * 2 + 1], q[half * 2 + 4], q[half * 2 + 5]);
      if (cplx) {
        const float4* qi = (const float4*)(Xi + inBase + (long)nn * 192 + kb);
        float4 i0 = qi[half * 2], i1 = qi[half * 2 + 1], i2 = qi[half * 2 + 4], i3 = qi[half * 2 + 5];
        bi = pack16(i0, i1, i2, i3);
        i0.x=-i0.x;i0.y=-i0.y;i0.z=-i0.z;i0.w=-i0.w; i1.x=-i1.x;i1.y=-i1.y;i1.z=-i1.z;i1.w=-i1.w;
        i2.x=-i2.x;i2.y=-i2.y;i2.z=-i2.z;i2.w=-i2.w; i3.x=-i3.x;i3.y=-i3.y;i3.z=-i3.z;i3.w=-i3.w;
        biN = pack16(i0, i1, i2, i3);
      }
      if (c6 + 1 < 6)
        __builtin_prefetch(bp + 32, 0, 1);
    } else {
#pragma unroll
      for (int e = 0; e < 16; ++e) {
        int j = kb + kmapA(e, half);
        long addr = inBase + (long)j * 192 + nn;
        float xr = Xr[addr];
        float xi = cplx ? Xi[addr] : 0.0f;
        br[e]  = (__bf16)xr;
        bi[e]  = (__bf16)xi;
        biN[e] = (__bf16)(-xi);
      }
    }
    accR = wmma_bf16(ar, br, accR);
    if (cplx) accR = wmma_bf16(ai, biN, accR);      // Re = Ar*Br - Ai*Bi
    if (wantIm) {
      accI = wmma_bf16(ai, br, accI);               // Im = Ai*Br (+ Ar*Bi)
      if (cplx) accI = wmma_bf16(ar, bi, accI);
    }
  }
#pragma unroll
  for (int v = 0; v < 8; ++v) {
    int m = blockIdx.y * 16 + half * 8 + v;
    long addr = outBase + (alongCols ? ((long)m * 192 + nn) : ((long)nn * 192 + m));
    Yr[addr] = accR[v] * scale;
    if (wantIm) Yi[addr] = accI[v] * scale;
  }
}

// ---------------------------------------------------------------------------
// 7x7 neighborhood attention, one thread per (b, head, pixel). HD=8, 49 taps.
// QKV layout: (B, 96, HW): rows 0..31 q, 32..63 k, 64..95 v (head-major).
// ---------------------------------------------------------------------------
__global__ void nat_attention_kernel(const float* __restrict__ QKV,
                                     const float* __restrict__ rpb,
                                     float* __restrict__ Out)
{
  int t = blockIdx.x * blockDim.x + threadIdx.x;
  if (t >= Bc * HEADSc * HWc) return;
  int hw = t % HWc;
  int r = t / HWc;
  int head = r % HEADSc;
  int b = r / HEADSc;
  int h = hw / Wc, w = hw % Wc;
  long base = (long)b * 96 * HWc;
  const float* qp = QKV + base + (long)(head * HDc) * HWc + hw;
  const float* kp = QKV + base + (long)(32 + head * HDc) * HWc;
  const float* vp = QKV + base + (long)(64 + head * HDc) * HWc;
  float qv[HDc];
#pragma unroll
  for (int d = 0; d < HDc; ++d) qv[d] = qp[(long)d * HWc];
  int sh = h - 3; sh = sh < 0 ? 0 : (sh > Hc - KSc ? Hc - KSc : sh);
  int sw = w - 3; sw = sw < 0 ? 0 : (sw > Wc - KSc ? Wc - KSc : sw);
  float logits[KSc * KSc];
  float mx = -1e30f;
  for (int i = 0; i < KSc; ++i) {
    int hh = sh + i, bih = hh - h + (KSc - 1);
    for (int j = 0; j < KSc; ++j) {
      int ww = sw + j, biw = ww - w + (KSc - 1);
      int off = hh * Wc + ww;
      float d0 = 0.f;
#pragma unroll
      for (int d = 0; d < HDc; ++d) d0 += qv[d] * kp[(long)d * HWc + off];
      float l = d0 * 0.35355339059327373f + rpb[head * 169 + bih * 13 + biw];
      logits[i * KSc + j] = l;
      mx = fmaxf(mx, l);
    }
  }
  float se = 0.f;
  for (int k = 0; k < KSc * KSc; ++k) { logits[k] = __expf(logits[k] - mx); se += logits[k]; }
  float inv = 1.f / se;
  float ov[HDc] = {0.f, 0.f, 0.f, 0.f, 0.f, 0.f, 0.f, 0.f};
  for (int i = 0; i < KSc; ++i) {
    int hh = sh + i;
    for (int j = 0; j < KSc; ++j) {
      int off = hh * Wc + (sw + j);
      float pw = logits[i * KSc + j] * inv;
#pragma unroll
      for (int d = 0; d < HDc; ++d) ov[d] += pw * vp[(long)d * HWc + off];
    }
  }
#pragma unroll
  for (int d = 0; d < HDc; ++d)
    Out[(long)b * 32 * HWc + (long)(head * HDc + d) * HWc + hw] = ov[d];
}

// ------------------------- pointwise kernels -------------------------------
__global__ void faf_mod_kernel(const float* __restrict__ Re, const float* __restrict__ Im,
                               const float* __restrict__ G, const float* __restrict__ T,
                               const float* __restrict__ Gi,
                               float* __restrict__ Or, float* __restrict__ Oi, long n)
{
  long i = (long)blockIdx.x * blockDim.x + threadIdx.x;
  if (i >= n) return;
  float re = Re[i], im = Im[i];
  float mag = sqrtf(re * re + im * im);
  float ang = atan2f(im, re);
  float yr = mag * G[i] + T[i];
  float yi = Gi[i] * (ang + 1.0f);
  float s, c; __sincosf(yi, &s, &c);
  Or[i] = yr * c; Oi[i] = yr * s;
}

__global__ void absang_kernel(const float* __restrict__ Re, const float* __restrict__ Im,
                              float* __restrict__ Mg, float* __restrict__ An, long n)
{
  long i = (long)blockIdx.x * blockDim.x + threadIdx.x;
  if (i >= n) return;
  float re = Re[i], im = Im[i];
  Mg[i] = sqrtf(re * re + im * im);
  An[i] = atan2f(im, re);
}

__global__ void polar_kernel(const float* __restrict__ Rm, const float* __restrict__ Am,
                             float* __restrict__ Or, float* __restrict__ Oi, long n)
{
  long i = (long)blockIdx.x * blockDim.x + threadIdx.x;
  if (i >= n) return;
  float s, c; __sincosf(Am[i], &s, &c);
  float r = Rm[i];
  Or[i] = r * c; Oi[i] = r * s;
}

__global__ void concat_kernel(const float* __restrict__ lo, const float* __restrict__ hi,
                              float* __restrict__ dst, long n)
{
  long i = (long)blockIdx.x * blockDim.x + threadIdx.x;
  if (i >= n) return;
  int hw = (int)(i % HWc);
  int c  = (int)((i / HWc) % DIMc);
  int b  = (int)(i / ((long)HWc * DIMc));
  dst[i] = (c < 32) ? lo[((long)b * 32 + c) * HWc + hw]
                    : hi[((long)b * 32 + (c - 32)) * HWc + hw];
}

__global__ void final_add_kernel(const float* __restrict__ sp, const float* __restrict__ fr,
                                 const float* __restrict__ l2, const float* __restrict__ x,
                                 float* __restrict__ out, long n)
{
  long i = (long)blockIdx.x * blockDim.x + threadIdx.x;
  if (i >= n) return;
  out[i] = sp[i] + fr[i] + l2[i] + x[i];
}

// ---------------------------------------------------------------------------
extern "C" void kernel_launch(void* const* d_in, const int* in_sizes, int n_in,
                              void* d_out, int out_size, void* d_ws, size_t ws_size,
                              hipStream_t stream)
{
  (void)in_sizes; (void)n_in; (void)out_size; (void)ws_size;
  // ---- inputs (setup_inputs dict order: x, then params in insertion order)
  const float* x        = (const float*)d_in[0];
  const float* ln1_g    = (const float*)d_in[1];
  const float* ln1_b    = (const float*)d_in[2];
  const float* ln2_g    = (const float*)d_in[3];
  const float* ln2_b    = (const float*)d_in[4];
  const float* faf_gr_w = (const float*)d_in[5];
  const float* faf_gr_b = (const float*)d_in[6];
  const float* faf_tr_w = (const float*)d_in[7];
  const float* faf_tr_b = (const float*)d_in[8];
  const float* faf_gi_w = (const float*)d_in[9];
  const float* faf_gi_b = (const float*)d_in[10];
  // d_in[11], d_in[12] = faf_ti_{w,b} (unused by reference)
  const float* n1_g     = (const float*)d_in[13];
  const float* n1_b     = (const float*)d_in[14];
  const float* n2_g     = (const float*)d_in[15];
  const float* n2_b     = (const float*)d_in[16];
  const float* qkv_w    = (const float*)d_in[17];
  const float* qkv_b    = (const float*)d_in[18];
  const float* proj_w   = (const float*)d_in[19];
  const float* proj_b   = (const float*)d_in[20];
  const float* rpb      = (const float*)d_in[21];
  const float* fc1_w    = (const float*)d_in[22];
  const float* fc1_b    = (const float*)d_in[23];
  const float* fc2_w    = (const float*)d_in[24];
  const float* fc2_b    = (const float*)d_in[25];
  const float* spw[3] = {(const float*)d_in[26], (const float*)d_in[28], (const float*)d_in[30]};
  const float* spb[3] = {(const float*)d_in[27], (const float*)d_in[29], (const float*)d_in[31]};
  const float* rew[3] = {(const float*)d_in[32], (const float*)d_in[34], (const float*)d_in[36]};
  const float* reb[3] = {(const float*)d_in[33], (const float*)d_in[35], (const float*)d_in[37]};
  const float* imw[3] = {(const float*)d_in[38], (const float*)d_in[40], (const float*)d_in[42]};
  const float* imb[3] = {(const float*)d_in[39], (const float*)d_in[41], (const float*)d_in[43]};
  float* out = (float*)d_out;

  // ---- workspace carve-up
  float* pool = (float*)d_ws;
  auto take = [&](long n) { float* p = pool; pool += n; return p; };
  float* L1   = take(N64);   // ln1 output (reused for sp-chain final)
  float* L2   = take(N64);   // ln2 output
  float* Abuf = take(N64);
  float* Bt   = take(N64);
  float* Cr   = take(N64);
  float* Ci   = take(N64);
  float* Dr   = take(N64);
  float* Di   = take(N64);
  float* natQ = take((long)Bc * 96 * HWc);
  float* natH = take((long)Bc * 128 * HWc);
  // aliases
  float* G = Abuf;  float* T = Abuf + N32;  float* Gi = Bt;  float* fafOut = Bt + N32;
  float* LNb = Cr;  float* ATT = Ci;        float* Ynat = Dr; float* Y2 = Di;

  const long BS64 = (long)DIMc * HWc, BS32 = (long)HALFc * HWc;
  const int  tokens = Bc * HWc;
  const int  CBIG = 1 << 30;
  dim3 blk(256);
  dim3 gln((tokens + 255) / 256);
  dim3 gconv32(288, 2, Bc), gconv64(288, 4, Bc);
  dim3 gdft32(96, 12, 1),   gdft64(192, 12, 1);
  int  gp32 = (int)((N32 + 255) / 256), gp64 = (int)((N64 + 255) / 256);

  // ===== 1. channel LN (ln1) =====
  channel_ln_kernel<<<gln, blk, 0, stream>>>(x, L1, ln1_g, ln1_b, 64, BS64, 0, BS64, tokens);

  // ===== 2. FAF branch (channels 0..31 of L1) =====
  conv3x3_wmma_kernel<32><<<gconv32, blk, 0, stream>>>(L1, BS64, faf_gr_w, faf_gr_b, G,  BS32);
  conv3x3_wmma_kernel<32><<<gconv32, blk, 0, stream>>>(L1, BS64, faf_tr_w, faf_tr_b, T,  BS32);
  conv3x3_wmma_kernel<32><<<gconv32, blk, 0, stream>>>(L1, BS64, faf_gi_w, faf_gi_b, Gi, BS32);
  // FFT2 (rows then cols), modulate, iFFT2 (real output)
  dft192_wmma_kernel<<<gdft32, blk, 0, stream>>>(L1, nullptr, Cr, Ci, 32, BS64, -1.f, 1.f, Bc * 32, 0);
  dft192_wmma_kernel<<<gdft32, blk, 0, stream>>>(Cr, Ci, Dr, Di, CBIG, 0, -1.f, 1.f, Bc * 32, 1);
  faf_mod_kernel<<<gp32, blk, 0, stream>>>(Dr, Di, G, T, Gi, Cr, Ci, N32);
  dft192_wmma_kernel<<<gdft32, blk, 0, stream>>>(Cr, Ci, Dr, Di, CBIG, 0, 1.f, 1.f / 192.f, Bc * 32, 0);
  dft192_wmma_kernel<<<gdft32, blk, 0, stream>>>(Dr, Di, fafOut, nullptr, CBIG, 0, 1.f, 1.f / 192.f, Bc * 32, 1);

  // ===== 3. NAT branch (channels 32..63 of L1) =====
  channel_ln_kernel<<<gln, blk, 0, stream>>>(L1, LNb, n1_g, n1_b, 32, BS64, 32, BS32, tokens);
  gemm_wmma_kernel<<<dim3(288, 6, Bc), blk, 0, stream>>>(qkv_w, 32, LNb, BS32, qkv_b,
      nullptr, 0, 0, natQ, (long)96 * HWc, 96, HWc, 32, 0);
  nat_attention_kernel<<<dim3((Bc * HEADSc * HWc + 255) / 256), blk, 0, stream>>>(natQ, rpb, ATT);
  gemm_wmma_kernel<<<dim3(288, 2, Bc), blk, 0, stream>>>(proj_w, 32, ATT, BS32, proj_b,
      L1, BS64, 32, Ynat, BS32, 32, HWc, 32, 0);
  channel_ln_kernel<<<gln, blk, 0, stream>>>(Ynat, LNb, n2_g, n2_b, 32, BS32, 0, BS32, tokens);
  gemm_wmma_kernel<<<dim3(288, 8, Bc), blk, 0, stream>>>(fc1_w, 32, LNb, BS32, fc1_b,
      nullptr, 0, 0, natH, (long)128 * HWc, 128, HWc, 32, 1 /*GELU*/);
  gemm_wmma_kernel<<<dim3(288, 2, Bc), blk, 0, stream>>>(fc2_w, 128, natH, (long)128 * HWc, fc2_b,
      Ynat, BS32, 0, Y2, BS32, 32, HWc, 128, 0);

  // ===== 4. concat + ln2 =====
  concat_kernel<<<gp64, blk, 0, stream>>>(fafOut, Y2, Abuf, N64);
  channel_ln_kernel<<<gln, blk, 0, stream>>>(Abuf, L2, ln2_g, ln2_b, 64, BS64, 0, BS64, tokens);

  // ===== 5. FFN: spatial conv chain =====
  conv3x3_wmma_kernel<64><<<gconv64, blk, 0, stream>>>(L2,   BS64, spw[0], spb[0], Abuf, BS64);
  conv3x3_wmma_kernel<64><<<gconv64, blk, 0, stream>>>(Abuf, BS64, spw[1], spb[1], Bt,   BS64);
  conv3x3_wmma_kernel<64><<<gconv64, blk, 0, stream>>>(Bt,   BS64, spw[2], spb[2], L1,   BS64);
  // ===== FFN: frequency path =====
  dft192_wmma_kernel<<<gdft64, blk, 0, stream>>>(L2, nullptr, Cr, Ci, CBIG, 0, -1.f, 1.f, Bc * 64, 0);
  dft192_wmma_kernel<<<gdft64, blk, 0, stream>>>(Cr, Ci, Dr, Di, CBIG, 0, -1.f, 1.f, Bc * 64, 1);
  absang_kernel<<<gp64, blk, 0, stream>>>(Dr, Di, Abuf, Bt, N64);
  conv3x3_wmma_kernel<64><<<gconv64, blk, 0, stream>>>(Abuf, BS64, rew[0], reb[0], Cr,   BS64);
  conv3x3_wmma_kernel<64><<<gconv64, blk, 0, stream>>>(Cr,   BS64, rew[1], reb[1], Abuf, BS64);
  conv3x3_wmma_kernel<64><<<gconv64, blk, 0, stream>>>(Abuf, BS64, rew[2], reb[2], Cr,   BS64);
  conv3x3_wmma_kernel<64><<<gconv64, blk, 0, stream>>>(Bt,   BS64, imw[0], imb[0], Ci,   BS64);
  conv3x3_wmma_kernel<64><<<gconv64, blk, 0, stream>>>(Ci,   BS64, imw[1], imb[1], Bt,   BS64);
  conv3x3_wmma_kernel<64><<<gconv64, blk, 0, stream>>>(Bt,   BS64, imw[2], imb[2], Ci,   BS64);
  polar_kernel<<<gp64, blk, 0, stream>>>(Cr, Ci, Dr, Di, N64);
  dft192_wmma_kernel<<<gdft64, blk, 0, stream>>>(Dr, Di, Abuf, Bt, CBIG, 0, 1.f, 1.f / 192.f, Bc * 64, 0);
  dft192_wmma_kernel<<<gdft64, blk, 0, stream>>>(Abuf, Bt, Cr, nullptr, CBIG, 0, 1.f, 1.f / 192.f, Bc * 64, 1);

  // ===== 6. out = sp + freq + ln2_out + shortcut =====
  final_add_kernel<<<gp64, blk, 0, stream>>>(L1, Cr, L2, x, out, N64);
}
